// QLSTMClassic_65481071397279
// MI455X (gfx1250) — matrix operand order repaired and checked by reading, not compiled, not run on hardware
//
#include <hip/hip_runtime.h>

// ---------------------------------------------------------------------------
// QLSTM for MI455X (gfx1250): bf16 WMMA (v_wmma_f32_16x16x32_bf16) GEMMs.
// Recurrent gate weights staged once into LDS (128KB/block, reused 512 steps
// by 4 waves via ds_load_b128); cell state register-resident for all steps;
// persistent kernel with one global barrier per step.
// T=512, B=64, D=512, H=512.
// ---------------------------------------------------------------------------

typedef __attribute__((ext_vector_type(16))) __bf16     v16bf;
typedef __attribute__((ext_vector_type(8)))  float      v8f;
typedef __attribute__((ext_vector_type(4)))  unsigned   v4u;

#define T_STEPS 512
#define BATCH   64
#define DIN     512
#define HID     512
#define KTOT    (DIN + HID)          // 1024
#define LDSROW  1032                 // 1024 + 8 ushort pad (bank decorrelation)
#define TBH     ((size_t)T_STEPS * BATCH * HID)   // 16,777,216

union BFrag {
    struct { v4u lo, hi; } u;
    v16bf v;
};

__device__ __forceinline__ v16bf load_frag_a(const unsigned short* row, int kbase, int hi) {
    // 16-bit A 16x32: lane holds row m=lane&15; VGPR0-3: K=kbase+hi*8..+7,
    // VGPR4-7: K=kbase+16+hi*8..+7  (two contiguous 16B chunks)
    BFrag f;
    f.u.lo = *(const v4u*)(row + kbase + hi * 8);
    f.u.hi = *(const v4u*)(row + kbase + 16 + hi * 8);
    return f.v;
}

__device__ __forceinline__ v16bf load_frag_b(const unsigned short* wrow, int kbase, int hi) {
    // 16-bit B 32x16: lane holds col n=lane&15; VGPR0-3: K=kbase+hi*8..+7,
    // VGPR4-7: K=kbase+16+hi*8..+7  (two contiguous 16B chunks)
    BFrag f;
    const unsigned short* p = wrow + kbase + hi * 8;
    f.u.lo = *(const v4u*)(p);
    f.u.hi = *(const v4u*)(p + 16);
    return f.v;
}

__device__ __forceinline__ v8f wmma_bf16(v16bf a, v16bf b, v8f c) {
    return __builtin_amdgcn_wmma_f32_16x16x32_bf16(
        /*neg_a=*/false, a, /*neg_b=*/false, b,
        /*c_mod=*/(short)0, c, /*reuse_a=*/false, /*reuse_b=*/false);
}

__device__ __forceinline__ unsigned short f2bf(float f) {
    unsigned u = __float_as_uint(f);
    unsigned r = u + 0x7FFFu + ((u >> 16) & 1u);   // round-to-nearest-even
    return (unsigned short)(r >> 16);
}

__device__ __forceinline__ float sigm(float x) { return 1.0f / (1.0f + __expf(-x)); }

// ---------------------------------------------------------------------------
// f32 -> bf16 conversion (grid-stride)
// ---------------------------------------------------------------------------
__global__ void cvt_f32_bf16(const float* __restrict__ src,
                             unsigned short* __restrict__ dst, long n) {
    long i = (long)blockIdx.x * blockDim.x + threadIdx.x;
    long stride = (long)gridDim.x * blockDim.x;
    for (; i < n; i += stride) dst[i] = f2bf(src[i]);
}

// ---------------------------------------------------------------------------
// zero h double-buffer + reset global barrier counter
// ---------------------------------------------------------------------------
__global__ void init_state(unsigned short* __restrict__ hb, unsigned* __restrict__ bar) {
    long i = (long)blockIdx.x * blockDim.x + threadIdx.x;
    long n = 2L * BATCH * HID;                     // both h buffers
    long stride = (long)gridDim.x * blockDim.x;
    for (; i < n; i += stride) hb[i] = 0;
    if (blockIdx.x == 0 && threadIdx.x == 0) *bar = 0u;
}

// ---------------------------------------------------------------------------
// Residual projection: res[t,b,:] = x @ W_r^T + b_r  -> written into d_out
// [32768,512] x [512,512], one 16x16 tile per wave, K=512 (16 WMMAs)
// ---------------------------------------------------------------------------
__global__ __launch_bounds__(256) void res_gemm(const unsigned short* __restrict__ xb,
                                                const unsigned short* __restrict__ wr,
                                                const float* __restrict__ br,
                                                float* __restrict__ out) {
    const int lane = threadIdx.x & 31;
    const int wave = threadIdx.x >> 5;
    const long tile = (long)blockIdx.x * 8 + wave;  // 65536 tiles
    const int rt = (int)(tile >> 5);                // row tile (of 2048)
    const int ct = (int)(tile & 31);                // col tile (of 32)
    const int rb = rt * 16, cb = ct * 16;
    const int l15 = lane & 15, hi = lane >> 4;
    const int j = cb + l15;

    const unsigned short* arow = xb + (size_t)(rb + l15) * DIN;
    const unsigned short* wrow = wr + (size_t)j * DIN;

    v8f acc = {0.f, 0.f, 0.f, 0.f, 0.f, 0.f, 0.f, 0.f};
#pragma unroll
    for (int kt = 0; kt < 16; ++kt) {
        const int kb = kt * 32;
        v16bf A = load_frag_a(arow, kb, hi);
        v16bf B = load_frag_b(wrow, kb, hi);
        acc = wmma_bf16(A, B, acc);
    }
    const float bias = br[j];
#pragma unroll
    for (int i = 0; i < 8; ++i) {
        const int r = rb + i + hi * 8;
        out[(size_t)r * HID + j] = acc[i] + bias;
    }
}

// ---------------------------------------------------------------------------
// Persistent recurrent kernel. 32 blocks x 4 waves = 128 waves.
// Block owns one 16-wide hidden tile (ct = blockIdx.x); its 4 waves cover the
// 4 batch tiles. Gate weights for the block's 16 output units (4 x 16 x 1024
// bf16 = 128KB + pad) staged in LDS once, reused every step (ds_load_b128).
// Cell state stays in registers for all 512 steps.
// ---------------------------------------------------------------------------
__device__ __forceinline__ void grid_sync(unsigned* bar, unsigned target) {
    __threadfence();
    __syncthreads();
    if (threadIdx.x == 0) {
        __hip_atomic_fetch_add(bar, 1u, __ATOMIC_ACQ_REL, __HIP_MEMORY_SCOPE_AGENT);
        while (__hip_atomic_load(bar, __ATOMIC_ACQUIRE, __HIP_MEMORY_SCOPE_AGENT) < target) {
            __builtin_amdgcn_s_sleep(2);
        }
    }
    __syncthreads();
}

__global__ __launch_bounds__(128, 1) void lstm_recurrent(
        const unsigned short* __restrict__ xb,   // [T*B, 512] bf16
        const unsigned short* __restrict__ Wg,   // [4][512][1024] bf16 (f,i,g,o)
        const float* __restrict__ bf_, const float* __restrict__ bi_,
        const float* __restrict__ bg_, const float* __restrict__ bo_,
        unsigned short* __restrict__ hb,         // [2][64][512] bf16 double buffer
        float* __restrict__ out,                 // stacked | hx | cx
        unsigned* __restrict__ bar) {
    extern __shared__ unsigned short lds[];      // [4*16][LDSROW] = 132,096 B

    const int lane = threadIdx.x & 31;
    const int rt = threadIdx.x >> 5;             // batch tile 0..3 (wave id)
    const int ct = blockIdx.x;                   // hidden tile 0..31
    const int rb = rt * 16, cb = ct * 16;
    const int l15 = lane & 15, hi = lane >> 4;
    const int hi8 = hi * 8;
    const int j = cb + l15;

    // --- stage this block's gate weights into LDS (once) -------------------
    // lds[(gate*16 + col)*LDSROW + k], K contiguous, row padded by 8 ushorts
    for (int idx = threadIdx.x; idx < 8192; idx += 128) {
        const int row  = idx >> 7;               // 0..63 = gate*16 + col
        const int gate = row >> 4;
        const int col  = row & 15;
        const int off  = (idx & 127) * 8;        // ushort offset within K row
        v4u d = *(const v4u*)(Wg + ((size_t)gate * HID + cb + col) * KTOT + off);
        *(v4u*)(&lds[(size_t)row * LDSROW + off]) = d;
    }
    __syncthreads();

    const unsigned short* wf_l = lds + (size_t)(0 * 16 + l15) * LDSROW;
    const unsigned short* wi_l = lds + (size_t)(1 * 16 + l15) * LDSROW;
    const unsigned short* wg_l = lds + (size_t)(2 * 16 + l15) * LDSROW;
    const unsigned short* wo_l = lds + (size_t)(3 * 16 + l15) * LDSROW;

    const float Bf = bf_[j], Bi = bi_[j], Bg = bg_[j], Bo = bo_[j];

    v8f cst = {0.f, 0.f, 0.f, 0.f, 0.f, 0.f, 0.f, 0.f};   // cell state, in regs

#pragma unroll 1
    for (int t = 0; t < T_STEPS; ++t) {
        const unsigned short* xrow = xb + ((size_t)t * BATCH + rb + l15) * DIN;
        const unsigned short* hrd  = hb + (size_t)(t & 1) * (BATCH * HID)
                                        + (size_t)(rb + l15) * HID;
        unsigned short*       hwr  = hb + (size_t)((t + 1) & 1) * (BATCH * HID);

        if (t + 1 < T_STEPS) {                   // prefetch next step's x row
            __builtin_prefetch(xrow + (size_t)BATCH * DIN, 0, 0);
            __builtin_prefetch(xrow + (size_t)BATCH * DIN + 256, 0, 0);
        }

        v8f af = {0.f,0.f,0.f,0.f,0.f,0.f,0.f,0.f};
        v8f ai = af, ag = af, ao = af;

#pragma unroll 8
        for (int kt = 0; kt < 32; ++kt) {
            const unsigned short* aptr = (kt < 16) ? xrow : hrd;
            const int kb = (kt & 15) * 32;
            v16bf A = load_frag_a(aptr, kb, hi);
            const int wk = kt * 32;
            af = wmma_bf16(A, load_frag_b(wf_l, wk, hi), af);
            ai = wmma_bf16(A, load_frag_b(wi_l, wk, hi), ai);
            ag = wmma_bf16(A, load_frag_b(wg_l, wk, hi), ag);
            ao = wmma_bf16(A, load_frag_b(wo_l, wk, hi), ao);
        }

#pragma unroll
        for (int i = 0; i < 8; ++i) {
            const int r = rb + i + hi8;
            const size_t oidx = ((size_t)t * BATCH + r) * HID + j;
            const float fg = sigm(af[i] + Bf);
            const float ig = sigm(ai[i] + Bi);
            const float gg = tanhf(ag[i] + Bg);
            const float og = sigm(ao[i] + Bo);
            const float cn = fg * cst[i] + ig * gg;
            cst[i] = cn;
            const float hn = og * tanhf(cn) + out[oidx];   // + residual (pre-written)
            out[oidx] = hn;
            hwr[(size_t)r * HID + j] = f2bf(hn);
            if (t == T_STEPS - 1) {
                out[TBH + (size_t)r * HID + j] = hn;                       // hx
                out[TBH + (size_t)BATCH * HID + (size_t)r * HID + j] = cn; // cx
            }
        }

        grid_sync(bar, 32u * (unsigned)(t + 1));
    }
}

// ---------------------------------------------------------------------------
// Launch
// ---------------------------------------------------------------------------
extern "C" void kernel_launch(void* const* d_in, const int* in_sizes, int n_in,
                              void* d_out, int out_size, void* d_ws, size_t ws_size,
                              hipStream_t stream) {
    const float* inputs = (const float*)d_in[0];
    const float* W_f    = (const float*)d_in[1];
    const float* b_f    = (const float*)d_in[2];
    const float* W_i    = (const float*)d_in[3];
    const float* b_i    = (const float*)d_in[4];
    const float* W_g    = (const float*)d_in[5];
    const float* b_g    = (const float*)d_in[6];
    const float* W_o    = (const float*)d_in[7];
    const float* b_o    = (const float*)d_in[8];
    const float* W_r    = (const float*)d_in[9];
    const float* b_r    = (const float*)d_in[10];
    float* out = (float*)d_out;

    // workspace layout (bytes), all 16B-aligned
    char* ws = (char*)d_ws;
    unsigned short* x_bf  = (unsigned short*)(ws);                       // 33,554,432 B
    unsigned short* Wg_bf = (unsigned short*)(ws + 33554432);            //  4,194,304 B
    unsigned short* Wr_bf = (unsigned short*)(ws + 33554432 + 4194304);  //    524,288 B
    unsigned short* h_bf  = (unsigned short*)(ws + 33554432 + 4194304 + 524288); // 131,072 B
    unsigned*       bar   = (unsigned*)(ws + 33554432 + 4194304 + 524288 + 131072);

    const long NX = (long)T_STEPS * BATCH * DIN;      // 16,777,216
    const long NW = (long)HID * KTOT;                 //    524,288 per gate
    const long NR = (long)HID * DIN;                  //    262,144

    cvt_f32_bf16<<<4096, 256, 0, stream>>>(inputs, x_bf, NX);
    cvt_f32_bf16<<<512, 256, 0, stream>>>(W_f, Wg_bf + 0 * NW, NW);
    cvt_f32_bf16<<<512, 256, 0, stream>>>(W_i, Wg_bf + 1 * NW, NW);
    cvt_f32_bf16<<<512, 256, 0, stream>>>(W_g, Wg_bf + 2 * NW, NW);
    cvt_f32_bf16<<<512, 256, 0, stream>>>(W_o, Wg_bf + 3 * NW, NW);
    cvt_f32_bf16<<<256, 256, 0, stream>>>(W_r, Wr_bf, NR);

    init_state<<<64, 256, 0, stream>>>(h_bf, bar);

    // residual projection into d_out stacked region: 65536 tiles / 8 waves
    res_gemm<<<8192, 256, 0, stream>>>(x_bf, Wr_bf, b_r, out);

    // persistent recurrence: 32 blocks x 128 threads, 132,096 B dynamic LDS
    const int lds_bytes = 64 * LDSROW * 2;            // 132,096
    (void)hipFuncSetAttribute((const void*)lstm_recurrent,
                              hipFuncAttributeMaxDynamicSharedMemorySize, lds_bytes);
    lstm_recurrent<<<32, 128, lds_bytes, stream>>>(x_bf, Wg_bf, b_f, b_i, b_g, b_o,
                                                   h_bf, out, bar);
}
